// Encoder_38998303047974
// MI455X (gfx1250) — compile-verified, matrix operand order checked by reading, not run on hardware
//
#include <hip/hip_runtime.h>

// ---------------- types ----------------
typedef __attribute__((ext_vector_type(16))) _Float16 v16h;
typedef __attribute__((ext_vector_type(8)))  _Float16 v8h;
typedef __attribute__((ext_vector_type(8)))  float    v8f;
typedef _Float16 h16;

#define NENT_TOT 8204
#define NENT_PAD 8208
#define NREL 8192
#define NABS 4096
#define TTOK 4096

// ---------------- device helpers ----------------
__device__ __forceinline__ v16h afrag_load(const h16* __restrict__ rowp, int kbase) {
  v8h lo = *(const v8h*)(rowp + kbase);
  v8h hi = *(const v8h*)(rowp + kbase + 16);
  v16h a;
#pragma unroll
  for (int i = 0; i < 8; ++i) { a[i] = lo[i]; a[i + 8] = hi[i]; }
  return a;
}

// Async global->LDS 16B copy (CDNA5 GLOBAL_LOAD_ASYNC_TO_LDS_B128, ASYNCcnt).
// LDS destination address = low 32 bits of the generic pointer.
__device__ __forceinline__ void async_ld_b128(void* lds_ptr, const void* gptr) {
  unsigned lds_off = (unsigned)(unsigned long long)lds_ptr;
  unsigned long long ga = (unsigned long long)gptr;
  asm volatile("global_load_async_to_lds_b128 %0, %1, off"
               :: "v"(lds_off), "v"(ga)
               : "memory");
}
__device__ __forceinline__ void wait_async() {
  asm volatile("s_wait_asynccnt 0x0" ::: "memory");
}

// ---------------- WMMA GEMM: out[M,N] = A[M,K](f16) x Bpacked[K,N](f16) (+bias +extra, relu, rowmask) ----------------
// Bp is pre-packed per 16x16-col x 32-k fragment: ((nb*(K/32)+kb)*32 + lane)*16 halves.
__global__ void k_gemm(const h16* __restrict__ A, const h16* __restrict__ Bp,
                       const float* __restrict__ bias, const float* __restrict__ extra,
                       const unsigned char* __restrict__ rowmask,
                       float* __restrict__ outF, h16* __restrict__ outH,
                       int N, int K, int relu, int Mstore) {
  extern __shared__ h16 As[];  // 16 x K
  const int tid = threadIdx.x;
  const int m0 = blockIdx.x * 16;
  const int kd8 = K >> 3;
  // stage A tile into LDS via async tensor-path loads (no VGPR round-trip)
  for (int i = tid; i < 2 * K; i += 256) {       // 16*K/8 chunks of 8 halves
    int row = i / kd8;
    int cc = (i % kd8) * 8;
    async_ld_b128(&As[row * K + cc], &A[(size_t)(m0 + row) * K + cc]);
  }
  wait_async();
  __syncthreads();
  const int w = tid >> 5, lane = tid & 31;
  const int n0 = blockIdx.y * 128 + w * 16;
  if (n0 >= N) return;
  const int nb = n0 >> 4;
  const int r = lane & 15;
  const int kb8 = (lane < 16) ? 0 : 8;
  const int kBlocks = K >> 5;
  v8f c = {};
  const h16* bptr = Bp + ((size_t)nb * kBlocks) * 512 + lane * 16;
  const h16* arow0 = &As[r * K + kb8];
  for (int kb = 0; kb < kBlocks; ++kb) {
    // unconditional near-scope prefetch of the B stream a few K-blocks ahead
    // (prefetch returns no data / no errors; OOB overrun stays inside d_ws)
    __builtin_prefetch(bptr + (size_t)(kb + 4) * 512, 0, 3);
    const h16* arow = arow0 + kb * 32;
    v8h lo = *(const v8h*)(arow);
    v8h hi = *(const v8h*)(arow + 16);
    v16h a;
#pragma unroll
    for (int i = 0; i < 8; ++i) { a[i] = lo[i]; a[i + 8] = hi[i]; }
    v16h b = *(const v16h*)(bptr + (size_t)kb * 512);
    c = __builtin_amdgcn_wmma_f32_16x16x32_f16(false, a, false, b, (short)0, c, false, false);
  }
  const int ncol = n0 + r;
  const int rbase = (lane < 16) ? 0 : 8;
  float bv = bias ? bias[ncol] : 0.f;
#pragma unroll
  for (int i = 0; i < 8; ++i) {
    int row = m0 + rbase + i;
    float v = c[i] + bv;
    if (extra) v += extra[(size_t)row * N + ncol];
    if (relu && v < 0.f) v = 0.f;
    if (rowmask && !rowmask[row]) v = 0.f;
    if (row < Mstore) {
      if (outF) outF[(size_t)row * N + ncol] = v;
      if (outH) outH[(size_t)row * N + ncol] = (h16)v;
    }
  }
}

// ---------------- pack f32 [K,N] row-major weights into WMMA B-fragment layout (f16) ----------------
__global__ void k_packB(const float* __restrict__ B, h16* __restrict__ Bp, int K, int N, int Kp) {
  int t = blockIdx.x * 256 + threadIdx.x;
  int kBlocks = Kp >> 5;
  int total = kBlocks * (N >> 4) * 32;
  if (t >= total) return;
  int lane = t & 31;
  int frag = t >> 5;
  int kb = frag % kBlocks;
  int nb = frag / kBlocks;
  int col = (nb << 4) + (lane & 15);
  int kbase = (kb << 5) + ((lane < 16) ? 0 : 8);
  v16h o;
#pragma unroll
  for (int i = 0; i < 8; ++i) {
    int k0 = kbase + i, k1 = kbase + 16 + i;
    o[i]     = (k0 < K) ? (h16)B[(size_t)k0 * N + col] : (h16)0.f;
    o[i + 8] = (k1 < K) ? (h16)B[(size_t)k1 * N + col] : (h16)0.f;
  }
  *(v16h*)(&Bp[((size_t)frag * 32 + lane) * 16]) = o;
}

// ---------------- featurization / gather kernels ----------------
__global__ void k_ent_build(const int* __restrict__ team, const int* __restrict__ hist,
                            const float* __restrict__ sp_oh, const float* __restrict__ ab_oh,
                            const float* __restrict__ it_oh,
                            const float* __restrict__ sp_emb, const float* __restrict__ ab_emb,
                            const float* __restrict__ it_emb, const float* __restrict__ ac_emb,
                            h16* __restrict__ A, float* __restrict__ emb, unsigned char* __restrict__ mask) {
  int row = blockIdx.x, tid = threadIdx.x;
  const int* e = nullptr;
  if (row < 12) e = team + row * 24;
  else if (row < NENT_TOT) e = hist + (size_t)(row - 12) * 24;
  if (!e) {
    for (int c = tid; c < 576; c += 64) A[(size_t)row * 576 + c] = (h16)0.f;
    for (int c = tid; c < 256; c += 64) emb[(size_t)row * 256 + c] = 0.f;
    if (tid == 0) mask[row] = 0;
    return;
  }
  int sp = e[0], ab = e[1], it = e[2];
  for (int c = tid; c < 576; c += 64) {
    float v;
    if (c < 128) v = sp_oh[sp * 128 + c];
    else if (c < 256) v = ab_oh[ab * 128 + (c - 128)];
    else if (c < 384) v = it_oh[it * 128 + (c - 256)];
    else if (c < 396) { int j = c - 384; v = (j < 4) ? ((e[18] == j) ? 1.f : 0.f) : ((e[19] == j - 4) ? 1.f : 0.f); }
    else if (c < 400) { int j = c - 396; v = (j == 0) ? e[20] / 1023.f : (j == 1) ? sp / 512.f : (j == 2) ? ab / 512.f : it / 512.f; }
    else if (c < 544) { int j = c - 400; v = (float)((e[7 + (j >> 4)] >> (j & 15)) & 1); }
    else { int j = c - 544; v = (float)((e[16 + (j >> 4)] >> (j & 15)) & 1); }
    A[(size_t)row * 576 + c] = (h16)v;
  }
  for (int c = tid; c < 256; c += 64) {
    float v = sp_emb[sp * 256 + c] + ab_emb[ab * 256 + c] + it_emb[it * 256 + c]
            + ac_emb[(size_t)e[3] * 256 + c] + ac_emb[(size_t)e[4] * 256 + c]
            + ac_emb[(size_t)e[5] * 256 + c] + ac_emb[(size_t)e[6] * 256 + c];
    emb[(size_t)row * 256 + c] = v;
  }
  if (tid == 0) mask[row] = (sp != 0 && sp != 1) ? 1 : 0;
}

__global__ void k_rel_build(const int* __restrict__ edges,
                            const float* __restrict__ ab_oh, const float* __restrict__ it_oh,
                            const float* __restrict__ it_emb, const float* __restrict__ ab_emb,
                            const float* __restrict__ ac_emb, const float* __restrict__ ef_emb,
                            h16* __restrict__ A, float* __restrict__ emb, unsigned char* __restrict__ mask) {
  int row = blockIdx.x, tid = threadIdx.x;
  const int* e = edges + (size_t)row * 12;
  int ab = e[0], it = e[1], act = e[2], src = e[3];
  for (int c = tid; c < 384; c += 64) {
    float v;
    if (c < 128) v = ab_oh[ab * 128 + c];
    else if (c < 256) v = it_oh[it * 128 + (c - 128)];
    else if (c < 320) { int j = c - 256; v = (float)((e[4 + (j >> 4)] >> (j & 15)) & 1); }
    else if (c < 352) { int j = c - 320; v = (float)((e[8 + (j >> 4)] >> (j & 15)) & 1); }
    else if (c < 368) { int j = c - 352; v = (e[11] == j) ? 1.f : 0.f; }
    else v = 0.f;
    A[(size_t)row * 384 + c] = (h16)v;
  }
  for (int c = tid; c < 256; c += 64) {
    emb[(size_t)row * 256 + c] = it_emb[it * 256 + c] + ab_emb[ab * 256 + c]
        + ac_emb[(size_t)act * 256 + c] + ef_emb[(size_t)src * 256 + c];
  }
  if (tid == 0) mask[row] = (e[10] > 0) ? 1 : 0;
}

__global__ void k_abs_max(const int* __restrict__ edges, int* __restrict__ outmax) {
  __shared__ int s0[256], s1[256];
  int tid = threadIdx.x;
  int m0 = -2147483647, m1 = -2147483647;
  for (int t = tid; t < NABS; t += 256) {
    int a = edges[t * 5], b = edges[t * 5 + 1];
    m0 = a > m0 ? a : m0; m1 = b > m1 ? b : m1;
  }
  s0[tid] = m0; s1[tid] = m1; __syncthreads();
  for (int st = 128; st > 0; st >>= 1) {
    if (tid < st) { s0[tid] = s0[tid] > s0[tid+st] ? s0[tid] : s0[tid+st];
                    s1[tid] = s1[tid] > s1[tid+st] ? s1[tid] : s1[tid+st]; }
    __syncthreads();
  }
  if (tid == 0) { outmax[0] = s0[0]; outmax[1] = s1[0]; }
}

__global__ void k_abs_build(const int* __restrict__ edges, const int* __restrict__ mx,
                            h16* __restrict__ A, unsigned char* __restrict__ mask) {
  int row = blockIdx.x, tid = threadIdx.x;
  const int* e = edges + (size_t)row * 5;
  int turn = e[0] > mx[0] ? e[0] - mx[0] : mx[0] - e[0];
  int req  = e[1] > mx[1] ? e[1] - mx[1] : mx[1] - e[1];
  for (int c = tid; c < 64; c += 64) {
    float v;
    if (c < 8) v = (e[4] == c) ? 1.f : 0.f;
    else if (c < 24) v = (float)((turn >> (c - 8)) & 1);
    else if (c < 40) v = (float)((req >> (c - 24)) & 1);
    else if (c < 56) v = (float)((e[2] >> (c - 40)) & 1);
    else v = 0.f;
    A[(size_t)row * 64 + c] = (h16)v;
  }
  if (tid == 0) mask[row] = (e[3] > 0) ? 1 : 0;
}

// scatter encoder f32 outputs into merge matrix (f16, 4096 x 1280)
__global__ void k_scatter(const float* __restrict__ src, h16* __restrict__ dst, int rows, int mode) {
  int t = blockIdx.x * 256 + threadIdx.x;
  if (t >= rows * 256) return;
  int r = t >> 8, c = t & 255;
  int row, col;
  if (mode == 2) { row = r; col = 1024 + c; }
  else { row = r >> 1; col = (mode ? 512 : 0) + (r & 1) * 256 + c; }
  dst[(size_t)row * 1280 + col] = (h16)src[(size_t)r * 256 + c];
}

__global__ void k_team_copy(const float* __restrict__ entOut, const unsigned char* __restrict__ maskE,
                            float* __restrict__ etx, unsigned char* __restrict__ etmask) {
  int row = blockIdx.x, c = threadIdx.x;
  etx[row * 256 + c] = (row < 12) ? entOut[row * 256 + c] : 0.f;
  if (c == 0) etmask[row] = (row < 12) ? maskE[row] : 0;
}

// ---------------- layernorm (one wave per 256-wide row), f32 in -> f16 out ----------------
__global__ void k_layernorm(const float* __restrict__ x, const float* __restrict__ s,
                            const float* __restrict__ b, h16* __restrict__ out) {
  int row = blockIdx.x, lane = threadIdx.x;
  float v[8], sum = 0.f;
#pragma unroll
  for (int i = 0; i < 8; ++i) { v[i] = x[(size_t)row * 256 + lane + i * 32]; sum += v[i]; }
#pragma unroll
  for (int m = 1; m < 32; m <<= 1) sum += __shfl_xor(sum, m, 32);
  float mean = sum * (1.f / 256.f);
  float vs = 0.f;
#pragma unroll
  for (int i = 0; i < 8; ++i) { float d = v[i] - mean; vs += d * d; }
#pragma unroll
  for (int m = 1; m < 32; m <<= 1) vs += __shfl_xor(vs, m, 32);
  float rstd = rsqrtf(vs * (1.f / 256.f) + 1e-6f);
#pragma unroll
  for (int i = 0; i < 8; ++i) {
    int c = lane + i * 32;
    out[(size_t)row * 256 + c] = (h16)((v[i] - mean) * rstd * s[c] + b[c]);
  }
}

// ---------------- flash attention: 16-query tile per wave, H=4 heads, DH=64 ----------------
#define FW 4
__global__ void k_flash(const h16* __restrict__ qkv, const unsigned char* __restrict__ kmask,
                        h16* __restrict__ outH, int nk) {
  __shared__ __align__(16) h16 shV[FW][32 * 64];
  __shared__ __align__(16) h16 shP[FW][16 * 32];
  int lane = threadIdx.x & 31, w = threadIdx.x >> 5;
  int wg = blockIdx.x * FW + w;
  int qt = wg >> 2, h = wg & 3;
  int r = lane & 15;
  int kb8 = (lane < 16) ? 0 : 8;
  int rbase = (lane < 16) ? 0 : 8;
  const h16* qrow = qkv + (size_t)(qt * 16 + r) * 768 + h * 64;
  v16h qa0 = afrag_load(qrow, kb8);
  v16h qa1 = afrag_load(qrow, 32 + kb8);
  float mprev[8], l[8];
  v8f o[4];
#pragma unroll
  for (int i = 0; i < 8; ++i) { mprev[i] = -1e30f; l[i] = 0.f; }
#pragma unroll
  for (int i = 0; i < 4; ++i) o[i] = (v8f){};
  h16* shPw = shP[w];
  h16* shVw = shV[w];
  int iters = (nk + 31) >> 5;
  for (int it = 0; it < iters; ++it) {
    int j0 = it * 32;
    // async-stage V tile (32 keys x 64) into LDS while computing S
#pragma unroll
    for (int i = 0; i < 8; ++i) {
      int ch = lane + i * 32;
      int k = ch >> 3;
      int d8 = (ch & 7) * 8;
      int j = j0 + k; int jj = (j < nk) ? j : 0;
      async_ld_b128(&shVw[k * 64 + d8], qkv + (size_t)jj * 768 + 512 + h * 64 + d8);
    }
    v8f s0, s1;
#pragma unroll
    for (int kb = 0; kb < 2; ++kb) {
      int j = j0 + kb * 16 + r;
      int jj = (j < nk) ? j : 0;
      const h16* krow = qkv + (size_t)jj * 768 + 256 + h * 64;
      v16h blo = afrag_load(krow, kb8);
      v16h bhi = afrag_load(krow, 32 + kb8);
      v8f cc = {};
      cc = __builtin_amdgcn_wmma_f32_16x16x32_f16(false, qa0, false, blo, (short)0, cc, false, false);
      cc = __builtin_amdgcn_wmma_f32_16x16x32_f16(false, qa1, false, bhi, (short)0, cc, false, false);
      bool valid = (j < nk) && (kmask ? (kmask[jj] != 0) : true);
      float cm = valid ? 0.f : -1e9f;
#pragma unroll
      for (int i = 0; i < 8; ++i) { float sv = cc[i] * 0.125f + cm; if (kb) s1[i] = sv; else s0[i] = sv; }
    }
    float mnew[8], alpha[8];
#pragma unroll
    for (int i = 0; i < 8; ++i) {
      float t = fmaxf(s0[i], s1[i]);
#pragma unroll
      for (int m = 1; m < 16; m <<= 1) t = fmaxf(t, __shfl_xor(t, m, 32));
      mnew[i] = fmaxf(mprev[i], t);
      alpha[i] = __expf(mprev[i] - mnew[i]);
      mprev[i] = mnew[i];
    }
#pragma unroll
    for (int i = 0; i < 8; ++i) {
      float p0 = __expf(s0[i] - mnew[i]);
      float p1 = __expf(s1[i] - mnew[i]);
      float rs = p0 + p1;
#pragma unroll
      for (int m = 1; m < 16; m <<= 1) rs += __shfl_xor(rs, m, 32);
      l[i] = l[i] * alpha[i] + rs;
      int row = rbase + i;
      shPw[row * 32 + r] = (h16)p0;
      shPw[row * 32 + 16 + r] = (h16)p1;
#pragma unroll
      for (int nbv = 0; nbv < 4; ++nbv) o[nbv][i] *= alpha[i];
    }
    wait_async();
    __syncthreads();
    v16h pf = afrag_load(&shPw[r * 32], kb8);
#pragma unroll
    for (int nbv = 0; nbv < 4; ++nbv) {
      v16h vf;
#pragma unroll
      for (int i = 0; i < 8; ++i) {
        vf[i]     = shVw[(kb8 + i) * 64 + nbv * 16 + r];
        vf[i + 8] = shVw[(16 + kb8 + i) * 64 + nbv * 16 + r];
      }
      o[nbv] = __builtin_amdgcn_wmma_f32_16x16x32_f16(false, pf, false, vf, (short)0, o[nbv], false, false);
    }
    __syncthreads();
  }
#pragma unroll
  for (int i = 0; i < 8; ++i) {
    float inv = (l[i] > 0.f) ? 1.f / l[i] : 0.f;
    int row = qt * 16 + rbase + i;
#pragma unroll
    for (int nbv = 0; nbv < 4; ++nbv)
      outH[(size_t)row * 256 + h * 64 + nbv * 16 + r] = (h16)(o[nbv][i] * inv);
  }
}

// ---------------- action head featurization ----------------
__global__ void k_act_build(const int* __restrict__ moveset, const float* __restrict__ mv_oh,
                            const float* __restrict__ ac_emb, const float* __restrict__ ctx_ent,
                            h16* __restrict__ A, float* __restrict__ emb, unsigned char* __restrict__ amask) {
  int row = blockIdx.x, tid = threadIdx.x;
  if (row >= 10) {
    for (int c = tid; c < 192; c += 64) A[(size_t)row * 192 + c] = (h16)0.f;
    for (int c = tid; c < 256; c += 64) emb[(size_t)row * 256 + c] = 0.f;
    if (tid == 0) amask[row] = 0;
    return;
  }
  const int* ms = moveset + row * 5;
  int aid = ms[0], mid = ms[1], m2 = ms[2], pp = ms[3], ei = ms[4];
  int ppc = pp < 0 ? 0 : (pp > 31 ? 31 : pp);
  for (int c = tid; c < 192; c += 64) {
    float v;
    if (c < 128) v = mv_oh[mid * 128 + c];
    else if (c < 130) v = (m2 == c - 128) ? 1.f : 0.f;
    else if (c < 162) v = (ppc == c - 130) ? 1.f : 0.f;
    else if (c == 162) v = (float)pp;
    else v = 0.f;
    A[(size_t)row * 192 + c] = (h16)v;
  }
  for (int c = tid; c < 256; c += 64)
    emb[(size_t)row * 256 + c] = ac_emb[(size_t)aid * 256 + c] + ctx_ent[ei * 256 + c];
  if (tid == 0) amask[row] = (aid > 3) ? 1 : 0;
}

// ---------------- host ----------------
static void gemm(const h16* A, const h16* Bp, const float* bias, const float* extra,
                 const unsigned char* rm, float* oF, h16* oH,
                 int M, int N, int K, int relu, int Mstore, hipStream_t s) {
  dim3 g(M / 16, (N + 127) / 128), b(256);
  k_gemm<<<g, b, (size_t)16 * K * 2, s>>>(A, Bp, bias, extra, rm, oF, oH, N, K, relu, Mstore);
}

extern "C" void kernel_launch(void* const* d_in, const int* in_sizes, int n_in,
                              void* d_out, int out_size, void* d_ws, size_t ws_size,
                              hipStream_t stream) {
  char* ws = (char*)d_ws;
  size_t off = 0;
  auto alloc = [&](size_t bytes) -> size_t { size_t r = off; off = (off + bytes + 255) & ~(size_t)255; return r; };

  const int* team = (const int*)d_in[0];
  const int* hist_ent = (const int*)d_in[1];
  const int* hist_rel = (const int*)d_in[2];
  const int* hist_abs = (const int*)d_in[3];
  const int* moveset = (const int*)d_in[4];
  const float* sp_oh = (const float*)d_in[6];
  const float* ab_oh = (const float*)d_in[7];
  const float* it_oh = (const float*)d_in[8];
  const float* mv_oh = (const float*)d_in[9];
  const float* sp_emb = (const float*)d_in[10];
  const float* ab_emb = (const float*)d_in[11];
  const float* it_emb = (const float*)d_in[12];
  const float* ac_emb = (const float*)d_in[13];
  const float* ef_emb = (const float*)d_in[14];

  // weight packing table: {input idx, K, N, Kpad}
  struct PB { int idx, K, N, Kp; size_t o; };
  PB pb[17] = {
    {15, 576, 256, 576, 0},  {17, 256, 256, 256, 0},  {19, 368, 256, 384, 0},
    {21, 256, 256, 256, 0},  {23, 56, 256, 64, 0},    {25, 256, 256, 256, 0},
    {27, 1280, 256, 1280, 0},{29, 163, 256, 192, 0},  {31, 256, 256, 256, 0},
    {35, 256, 768, 256, 0},  {36, 256, 256, 256, 0},  {39, 256, 1024, 256, 0},
    {41, 1024, 256, 1024, 0},{45, 256, 768, 256, 0},  {46, 256, 256, 256, 0},
    {49, 256, 1024, 256, 0}, {51, 1024, 256, 1024, 0}
  };
  enum { ENT_ENC, ENT_MLP, REL_ENC, REL_MLP, ABS_ENC, ABS_MLP, MERGE, ACT_ENC, ACT_MLP,
         TS_QKV, TS_WO, TS_W1, TS_W2, ET_QKV, ET_WO, ET_W1, ET_W2 };
  for (int i = 0; i < 17; ++i) {
    pb[i].o = alloc((size_t)pb[i].Kp * pb[i].N * 2);
    int total = (pb[i].Kp / 32) * (pb[i].N / 16) * 32;
    k_packB<<<(total + 255) / 256, 256, 0, stream>>>((const float*)d_in[pb[i].idx],
                                                     (h16*)(ws + pb[i].o), pb[i].K, pb[i].N, pb[i].Kp);
  }
  auto BP = [&](int i) { return (const h16*)(ws + pb[i].o); };

  // buffers (with reuse of the big pool regions across phases)
  size_t oMERGEA = alloc((size_t)TTOK * 1280 * 2);      // also reused as FFN f16 buffer
  size_t oTS     = alloc((size_t)TTOK * 256 * 4);
  size_t oTSMASK = alloc(TTOK);
  size_t oMASKE  = alloc(NENT_PAD);
  size_t oMASKR  = alloc(NREL);
  size_t oETX    = alloc(16 * 256 * 4);
  size_t oETMASK = alloc(16);
  size_t oCTXENT = alloc(16 * 256 * 4);
  size_t oABSMAX = alloc(2 * 4);
  size_t oACTMSK = alloc(16);
  size_t oA   = alloc((size_t)NENT_PAD * 576 * 2);      // A staging / QKV f16
  size_t oEMB = alloc((size_t)NENT_PAD * 256 * 4);      // emb f32 / attn f16
  size_t oH1  = alloc((size_t)NENT_PAD * 256 * 2);      // hidden f16 / LN outputs
  size_t oOUT = alloc((size_t)NENT_PAD * 256 * 4);      // f32 encoder out / residual x2
  (void)ws_size; (void)n_in; (void)in_sizes; (void)out_size;

  h16* MERGEA = (h16*)(ws + oMERGEA);
  float* TS = (float*)(ws + oTS);
  unsigned char* TSMASK = (unsigned char*)(ws + oTSMASK);
  unsigned char* MASKE = (unsigned char*)(ws + oMASKE);
  unsigned char* MASKR = (unsigned char*)(ws + oMASKR);
  float* ETX = (float*)(ws + oETX);
  unsigned char* ETMASK = (unsigned char*)(ws + oETMASK);
  float* CTXENT = (float*)(ws + oCTXENT);
  int* ABSMAX = (int*)(ws + oABSMAX);
  unsigned char* ACTMSK = (unsigned char*)(ws + oACTMSK);
  h16* Abuf = (h16*)(ws + oA);
  float* EMB = (float*)(ws + oEMB);
  h16* H1 = (h16*)(ws + oH1);
  float* OUT = (float*)(ws + oOUT);

  // ---- entity encoder ----
  k_ent_build<<<NENT_PAD, 64, 0, stream>>>(team, hist_ent, sp_oh, ab_oh, it_oh,
                                           sp_emb, ab_emb, it_emb, ac_emb, Abuf, EMB, MASKE);
  gemm(Abuf, BP(ENT_ENC), (const float*)d_in[16], EMB, nullptr, nullptr, H1,
       NENT_PAD, 256, 576, 1, NENT_PAD, stream);
  gemm(H1, BP(ENT_MLP), (const float*)d_in[18], nullptr, MASKE, OUT, nullptr,
       NENT_PAD, 256, 256, 0, NENT_PAD, stream);
  k_scatter<<<NREL, 256, 0, stream>>>(OUT + 12 * 256, MERGEA, NREL, 0);
  k_team_copy<<<16, 256, 0, stream>>>(OUT, MASKE, ETX, ETMASK);

  // ---- relation encoder ----
  k_rel_build<<<NREL, 64, 0, stream>>>(hist_rel, ab_oh, it_oh, it_emb, ab_emb, ac_emb, ef_emb,
                                       Abuf, EMB, MASKR);
  gemm(Abuf, BP(REL_ENC), (const float*)d_in[20], EMB, nullptr, nullptr, H1,
       NREL, 256, 384, 1, NREL, stream);
  gemm(H1, BP(REL_MLP), (const float*)d_in[22], nullptr, MASKR, OUT, nullptr,
       NREL, 256, 256, 0, NREL, stream);
  k_scatter<<<NREL, 256, 0, stream>>>(OUT, MERGEA, NREL, 1);

  // ---- absolute-edge encoder ----
  k_abs_max<<<1, 256, 0, stream>>>(hist_abs, ABSMAX);
  k_abs_build<<<NABS, 64, 0, stream>>>(hist_abs, ABSMAX, Abuf, TSMASK);
  gemm(Abuf, BP(ABS_ENC), (const float*)d_in[24], nullptr, nullptr, nullptr, H1,
       NABS, 256, 64, 1, NABS, stream);
  gemm(H1, BP(ABS_MLP), (const float*)d_in[26], nullptr, TSMASK, OUT, nullptr,
       NABS, 256, 256, 0, NABS, stream);
  k_scatter<<<NABS, 256, 0, stream>>>(OUT, MERGEA, NABS, 2);

  // ---- merge -> ts tokens ----
  gemm(MERGEA, BP(MERGE), (const float*)d_in[28], nullptr, TSMASK, TS, nullptr,
       TTOK, 256, 1280, 0, TTOK, stream);

  // ---- ts transformer ----
  h16* HLN = H1;
  h16* QKV = Abuf;
  h16* ATTN = (h16*)EMB;
  float* X2 = OUT;
  h16* H2 = H1;
  h16* FF = (h16*)MERGEA;
  k_layernorm<<<TTOK, 32, 0, stream>>>(TS, (const float*)d_in[33], (const float*)d_in[34], HLN);
  gemm(HLN, BP(TS_QKV), nullptr, nullptr, nullptr, nullptr, QKV, TTOK, 768, 256, 0, TTOK, stream);
  k_flash<<<TTOK / 16, FW * 32, 0, stream>>>(QKV, TSMASK, ATTN, TTOK);
  gemm(ATTN, BP(TS_WO), nullptr, TS, nullptr, X2, nullptr, TTOK, 256, 256, 0, TTOK, stream);
  k_layernorm<<<TTOK, 32, 0, stream>>>(X2, (const float*)d_in[37], (const float*)d_in[38], H2);
  gemm(H2, BP(TS_W1), (const float*)d_in[40], nullptr, nullptr, nullptr, FF,
       TTOK, 1024, 256, 1, TTOK, stream);
  gemm(FF, BP(TS_W2), (const float*)d_in[42], X2, TSMASK, (float*)d_out + 2560, nullptr,
       TTOK, 256, 1024, 0, TTOK, stream);

  // ---- entity transformer (n=16 padded, 12 valid) ----
  k_layernorm<<<16, 32, 0, stream>>>(ETX, (const float*)d_in[43], (const float*)d_in[44], HLN);
  gemm(HLN, BP(ET_QKV), nullptr, nullptr, nullptr, nullptr, QKV, 16, 768, 256, 0, 16, stream);
  k_flash<<<1, FW * 32, 0, stream>>>(QKV, ETMASK, ATTN, 16);
  gemm(ATTN, BP(ET_WO), nullptr, ETX, nullptr, X2, nullptr, 16, 256, 256, 0, 16, stream);
  k_layernorm<<<16, 32, 0, stream>>>(X2, (const float*)d_in[47], (const float*)d_in[48], H2);
  gemm(H2, BP(ET_W1), (const float*)d_in[50], nullptr, nullptr, nullptr, FF, 16, 1024, 256, 1, 16, stream);
  gemm(FF, BP(ET_W2), (const float*)d_in[52], X2, ETMASK, CTXENT, nullptr, 16, 256, 1024, 0, 16, stream);

  // ---- action head ----
  k_act_build<<<16, 64, 0, stream>>>(moveset, mv_oh, ac_emb, CTXENT, Abuf, EMB, ACTMSK);
  gemm(Abuf, BP(ACT_ENC), (const float*)d_in[30], EMB, nullptr, nullptr, H1, 16, 256, 192, 1, 16, stream);
  gemm(H1, BP(ACT_MLP), (const float*)d_in[32], nullptr, ACTMSK, (float*)d_out, nullptr,
       16, 256, 256, 0, 10, stream);
}
